// Model_63702954934509
// MI455X (gfx1250) — compile-verified, matrix operand order checked by reading, not run on hardware
//
#include <hip/hip_runtime.h>
#include <vector>
#include <cstddef>
#include <cstdint>

typedef __bf16 bf16;
typedef __attribute__((ext_vector_type(16))) __bf16 v16bf;
typedef __attribute__((ext_vector_type(8)))  __bf16 v8bf;
typedef __attribute__((ext_vector_type(8)))  float  v8f;

static inline int ceil_div(int a, int b){ return (a + b - 1) / b; }
static inline int align_up(int x, int a){ return ceil_div(x, a) * a; }

// =========================== device kernels ===========================

// f32 [N,C] -> bf16 [Nrows, Cpad]; rows >= N and cols >= C are zero (zero row
// at index >= N serves as the gather sentinel target).
__global__ void k_to_bf16(const float* __restrict__ x, bf16* __restrict__ y,
                          int N, int C, int Nrows, int Cpad){
  int t = blockIdx.x * blockDim.x + threadIdx.x;
  int total = Nrows * Cpad;
  if (t >= total) return;
  int r = t / Cpad, c = t - r * Cpad;
  float v = (r < N && c < C) ? x[(size_t)r * C + c] : 0.0f;
  y[t] = (bf16)v;
}

// W f32 [K,Cin,Cout] -> Wt bf16 [K, CoutPad, CinPad] (transposed, zero padded)
__global__ void k_prep_w(const float* __restrict__ w, bf16* __restrict__ wt,
                         int K, int Cin, int Cout, int CinPad, int CoutPad){
  int t = blockIdx.x * blockDim.x + threadIdx.x;
  int per = CoutPad * CinPad;
  if (t >= K * per) return;
  int k = t / per; int rem = t - k * per;
  int o = rem / CinPad; int c = rem - o * CinPad;
  float v = (c < Cin && o < Cout) ? w[((size_t)k * Cin + c) * Cout + o] : 0.0f;
  wt[t] = (bf16)v;
}

// out[N_out, CoutPad] = sum_k gather(xb, nbr_k) @ Wt_k  via WMMA bf16.
// One wave per 16x16 output tile; K offsets x CinPad/32 WMMA accumulations.
__global__ void k_wmma_conv(const bf16* __restrict__ xb, int CinPad,
                            const int* __restrict__ nbr, int K,
                            const bf16* __restrict__ wt, int CoutPad,
                            float* __restrict__ out,
                            int N_out, int N_in, int zrow, int totalTiles){
  int wave = (blockIdx.x * blockDim.x + threadIdx.x) >> 5;
  if (wave >= totalTiles) return;           // uniform per-wave exit
  int lane = threadIdx.x & 31;
  int tilesC = CoutPad >> 4;
  int tn = wave / tilesC;
  int tc = wave - tn * tilesC;
  int mrow = lane & 15;
  int hf = lane >> 4;
  int r = tn * 16 + mrow;
  int cA0 = hf * 8;
  v8f acc = {};
  const bf16* wcol = wt + (size_t)(tc * 16 + mrow) * CinPad;
  size_t wstride = (size_t)CoutPad * CinPad;
  for (int k = 0; k < K; ++k){
    int idx;
    if (nbr) idx = (r < N_out) ? nbr[(size_t)k * N_out + r] : N_in;
    else     idx = r;
    if ((unsigned)idx >= (unsigned)N_in) idx = zrow;   // sentinel -> zero row
    const bf16* arow = xb + (size_t)idx * CinPad;
    const bf16* brow = wcol + (size_t)k * wstride;
    for (int c0 = 0; c0 < CinPad; c0 += 32){
      const bf16* ap = arow + c0 + cA0;
      const bf16* bp = brow + c0 + cA0;
      v8bf a0 = *(const v8bf*)(ap);
      v8bf a1 = *(const v8bf*)(ap + 16);
      v8bf b0 = *(const v8bf*)(bp);
      v8bf b1 = *(const v8bf*)(bp + 16);
      v16bf A = __builtin_shufflevector(a0, a1, 0,1,2,3,4,5,6,7,8,9,10,11,12,13,14,15);
      v16bf B = __builtin_shufflevector(b0, b1, 0,1,2,3,4,5,6,7,8,9,10,11,12,13,14,15);
#if defined(__HIP_DEVICE_COMPILE__)
      acc = __builtin_amdgcn_wmma_f32_16x16x32_bf16(false, A, false, B,
                                                    (short)0, acc, false, false);
#endif
    }
  }
  int col = tc * 16 + mrow;
#pragma unroll
  for (int i = 0; i < 8; ++i){
    int row = tn * 16 + hf * 8 + i;
    if (row < N_out) out[(size_t)row * CoutPad + col] = acc[i];
  }
}

// training-mode BN: per-channel mean/var over N, fused into scale/shift
__global__ void k_bn_stats(const float* __restrict__ x, int N, int C,
                           const float* __restrict__ g, const float* __restrict__ b,
                           float* __restrict__ scale, float* __restrict__ shift){
  __shared__ float ss[256];
  __shared__ float ss2[256];
  int c = blockIdx.x;
  float s = 0.f, s2 = 0.f;
  for (int i = threadIdx.x; i < N; i += blockDim.x){
    float v = x[(size_t)i * C + c];
    s += v; s2 += v * v;
  }
  ss[threadIdx.x] = s; ss2[threadIdx.x] = s2;
  __syncthreads();
  for (int st = 128; st > 0; st >>= 1){
    if ((int)threadIdx.x < st){
      ss[threadIdx.x]  += ss[threadIdx.x + st];
      ss2[threadIdx.x] += ss2[threadIdx.x + st];
    }
    __syncthreads();
  }
  if (threadIdx.x == 0){
    float m  = ss[0] / (float)N;
    float vr = ss2[0] / (float)N - m * m;
    float sc = g[c] * rsqrtf(vr + 1e-5f);
    scale[c] = sc;
    shift[c] = b[c] - m * sc;
  }
}

__global__ void k_bn_apply(const float* __restrict__ x, float* __restrict__ y,
                           int N, int C,
                           const float* __restrict__ scale, const float* __restrict__ shift,
                           const float* __restrict__ res, int relu){
  int t = blockIdx.x * blockDim.x + threadIdx.x;
  if (t >= N * C) return;
  int c = t % C;
  float v = x[t] * scale[c] + shift[c];
  if (res) v += res[t];
  if (relu) v = fmaxf(v, 0.f);
  y[t] = v;
}

__global__ void k_sigmul(float* __restrict__ x, const float* __restrict__ s, int n){
  int t = blockIdx.x * blockDim.x + threadIdx.x;
  if (t < n) x[t] *= 1.f / (1.f + expf(-s[t]));
}

__global__ void k_copy_cols(float* __restrict__ dst, int Cdst, int colOff,
                            const float* __restrict__ src, int Csrc, int N){
  int t = blockIdx.x * blockDim.x + threadIdx.x;
  if (t >= N * Csrc) return;
  int r = t / Csrc, c = t - r * Csrc;
  dst[(size_t)r * Cdst + colOff + c] = src[t];
}

// synthetic neighbor map for sparse_conv_tr: nbr[o][n] = (offid[n]==o)?parent[n]:BIG
__global__ void k_upnbr(const int* __restrict__ parent, const int* __restrict__ offid,
                        int N, int* __restrict__ nbr){
  int i = blockIdx.x * blockDim.x + threadIdx.x;
  if (i >= N) return;
  int o = offid[i], p = parent[i];
#pragma unroll
  for (int k = 0; k < 8; ++k)
    nbr[(size_t)k * N + i] = (k == o) ? p : 0x7FFFFFFF;
}

// d_out = [logits N*20 | fea N*96]
__global__ void k_finalout(const float* __restrict__ lin, int ldLin,
                           const float* __restrict__ bias,
                           const float* __restrict__ fea,
                           float* __restrict__ out, int N){
  int t = blockIdx.x * blockDim.x + threadIdx.x;
  int nlog = N * 20;
  int total = nlog + N * 96;
  if (t >= total) return;
  if (t < nlog){
    int r = t / 20, c = t - r * 20;
    out[t] = lin[(size_t)r * ldLin + c] + bias[c];
  } else {
    out[t] = fea[t - nlog];
  }
}

// =========================== host side ===========================

struct WS {
  char* base; size_t cap; size_t off;
  void* alloc(size_t bytes){
    size_t a = (off + 255) & ~(size_t)255;
    off = a + bytes;
    return base + a;
  }
};
static float* allocF(WS& w, size_t n){ return (float*)w.alloc(n * sizeof(float)); }

struct Reader {
  void* const* in; const int* sz; int i;
  const float* F(){ return (const float*)in[i++]; }
  const int*   I(){ return (const int*)in[i++]; }
};
struct BNp { const float* g; const float* b; };
struct Blk { const float* w1; BNp bn1; const float* w2; BNp bn2;
             const float* wd; BNp bnd; int cin; int cout; };

static BNp rdBN(Reader& r){ BNp p; p.g = r.F(); p.b = r.F(); return p; }
static Blk rdBlk(Reader& r, int ci, int co){
  Blk b; b.cin = ci; b.cout = co;
  b.w1 = r.F(); b.bn1 = rdBN(r); b.w2 = r.F(); b.bn2 = rdBN(r);
  if (ci != co){ b.wd = r.F(); b.bnd = rdBN(r); } else { b.wd = nullptr; b.bnd = BNp{nullptr,nullptr}; }
  return b;
}
static void rdBlocks(Reader& r, std::vector<Blk>& v, int cin, int cout, int n){
  for (int i = 0; i < n; ++i) v.push_back(rdBlk(r, i == 0 ? cin : cout, cout));
}
struct Enc {
  const float* w0; BNp bn0; const float* w1; BNp bn1;
  std::vector<Blk> b1;
  const float* w2; BNp bn2; std::vector<Blk> b2;
  const float* w3; BNp bn3; std::vector<Blk> b3;
  const float* w4; BNp bn4; std::vector<Blk> b4;
};
static Enc rdEnc(Reader& r){
  Enc e;
  e.w0 = r.F(); e.bn0 = rdBN(r);
  e.w1 = r.F(); e.bn1 = rdBN(r);
  rdBlocks(r, e.b1, 32, 32, 2);
  e.w2 = r.F(); e.bn2 = rdBN(r);
  rdBlocks(r, e.b2, 32, 64, 3);
  e.w3 = r.F(); e.bn3 = rdBN(r);
  rdBlocks(r, e.b3, 64, 128, 4);
  e.w4 = r.F(); e.bn4 = rdBN(r);
  rdBlocks(r, e.b4, 128, 256, 6);
  return e;
}
struct Dec {
  const float* wtr4; BNp bntr4; std::vector<Blk> b5;
  const float* wtr5; BNp bntr5; std::vector<Blk> b6;
  const float* wtr6; BNp bntr6; std::vector<Blk> b7;
  const float* wtr7; BNp bntr7; std::vector<Blk> b8;
  const float* wf; const float* bf;
};
static Dec rdDec(Reader& r){
  Dec d;
  d.wtr4 = r.F(); d.bntr4 = rdBN(r); rdBlocks(r, d.b5, 384, 256, 2);
  d.wtr5 = r.F(); d.bntr5 = rdBN(r); rdBlocks(r, d.b6, 192, 128, 2);
  d.wtr6 = r.F(); d.bntr6 = rdBN(r); rdBlocks(r, d.b7, 128, 96, 2);
  d.wtr7 = r.F(); d.bntr7 = rdBN(r); rdBlocks(r, d.b8, 128, 96, 2);
  d.wf = r.F(); d.bf = r.F();
  return d;
}

// gather-GEMM: outRaw [N_out, CoutPad] = sum_k gather(x, nbr_k) @ W_k
static void conv_op(WS& ws, hipStream_t st,
                    const float* x, int N_in, int Cin,
                    const int* nbr, int K, int N_out,
                    const float* w, int Cout, int CoutPad,
                    float* outRaw){
  int CinPad = align_up(Cin, 32);
  int zrow = align_up(N_in, 16);
  int NrowsIn = zrow + 16;
  size_t mk = ws.off;
  bf16* xb = (bf16*)ws.alloc((size_t)NrowsIn * CinPad * sizeof(bf16));
  int tot1 = NrowsIn * CinPad;
  k_to_bf16<<<dim3(ceil_div(tot1, 256)), dim3(256), 0, st>>>(x, xb, N_in, Cin, NrowsIn, CinPad);
  bf16* wt = (bf16*)ws.alloc((size_t)K * CoutPad * CinPad * sizeof(bf16));
  int tot2 = K * CoutPad * CinPad;
  k_prep_w<<<dim3(ceil_div(tot2, 256)), dim3(256), 0, st>>>(w, wt, K, Cin, Cout, CinPad, CoutPad);
  int totalTiles = ceil_div(N_out, 16) * (CoutPad / 16);
  int blocks = ceil_div(totalTiles * 32, 256);
  k_wmma_conv<<<dim3(blocks), dim3(256), 0, st>>>(xb, CinPad, nbr, K, wt, CoutPad,
                                                  outRaw, N_out, N_in, zrow, totalTiles);
  ws.off = mk;   // stream-ordered reuse of staging is safe
}

static void bn_op(WS& ws, hipStream_t st, const float* x, float* y, int N, int C,
                  const BNp& p, const float* res, int relu){
  size_t mk = ws.off;
  float* sc = allocF(ws, C);
  float* sh = allocF(ws, C);
  k_bn_stats<<<dim3(C), dim3(256), 0, st>>>(x, N, C, p.g, p.b, sc, sh);
  k_bn_apply<<<dim3(ceil_div(N * C, 256)), dim3(256), 0, st>>>(x, y, N, C, sc, sh, res, relu);
  ws.off = mk;
}

static void basic_block(WS& ws, hipStream_t st, float* out,
                        const float* x, int N, const int* nbr, const Blk& b){
  size_t mk = ws.off;
  int co = b.cout;
  float* h = allocF(ws, (size_t)N * co);
  conv_op(ws, st, x, N, b.cin, nbr, 27, N, b.w1, co, co, h);
  bn_op(ws, st, h, h, N, co, b.bn1, nullptr, 1);
  float* h2 = allocF(ws, (size_t)N * co);
  conv_op(ws, st, h, N, co, nbr, 27, N, b.w2, co, co, h2);
  const float* r = x;
  if (b.wd){
    float* rb = allocF(ws, (size_t)N * co);
    conv_op(ws, st, x, N, b.cin, nullptr, 1, N, b.wd, co, co, rb);
    bn_op(ws, st, rb, rb, N, co, b.bnd, nullptr, 0);
    r = rb;
  }
  bn_op(ws, st, h2, out, N, co, b.bn2, r, 1);   // out = relu(bn2(h2) + r)
  ws.off = mk;
}

static void run_blocks(WS& ws, hipStream_t st, float* out,
                       const float* x, int N, const int* nbr, const std::vector<Blk>& bs){
  size_t mk = ws.off;
  int co = bs[0].cout;
  int n = (int)bs.size();
  float* p0 = nullptr; float* p1 = nullptr;
  if (n > 1){ p0 = allocF(ws, (size_t)N * co); p1 = allocF(ws, (size_t)N * co); }
  const float* cur = x;
  for (int i = 0; i < n; ++i){
    float* dst = (i == n - 1) ? out : ((i & 1) ? p1 : p0);
    basic_block(ws, st, dst, cur, N, nbr, bs[i]);
    cur = dst;
  }
  ws.off = mk;
}

struct EncOut { float* p1; float* b1; float* b2; float* b3; float* b4; };

static EncOut run_encoder(WS& ws, hipStream_t st, const float* feats, const Enc& e,
                          int N0, int N1, int N2, int N3, int N4,
                          const int* k5l0,
                          const int* d01, const int* d12, const int* d23, const int* d34,
                          const int* k3l1, const int* k3l2, const int* k3l3, const int* k3l4,
                          const float* g1, const float* g2, const float* g3){
  EncOut o;
  o.p1 = allocF(ws, (size_t)N0 * 32);
  conv_op(ws, st, feats, N0, 3, k5l0, 125, N0, e.w0, 32, 32, o.p1);
  bn_op(ws, st, o.p1, o.p1, N0, 32, e.bn0, nullptr, 1);

  o.b1 = allocF(ws, (size_t)N1 * 32);
  { size_t mk = ws.off;
    float* t = allocF(ws, (size_t)N1 * 32);
    conv_op(ws, st, o.p1, N0, 32, d01, 8, N1, e.w1, 32, 32, t);
    bn_op(ws, st, t, t, N1, 32, e.bn1, nullptr, 1);
    run_blocks(ws, st, o.b1, t, N1, k3l1, e.b1);
    ws.off = mk; }
  if (g1) k_sigmul<<<dim3(ceil_div(N1 * 32, 256)), dim3(256), 0, st>>>(o.b1, g1, N1 * 32);

  o.b2 = allocF(ws, (size_t)N2 * 64);
  { size_t mk = ws.off;
    float* t = allocF(ws, (size_t)N2 * 32);
    conv_op(ws, st, o.b1, N1, 32, d12, 8, N2, e.w2, 32, 32, t);
    bn_op(ws, st, t, t, N2, 32, e.bn2, nullptr, 1);
    run_blocks(ws, st, o.b2, t, N2, k3l2, e.b2);
    ws.off = mk; }
  if (g2) k_sigmul<<<dim3(ceil_div(N2 * 64, 256)), dim3(256), 0, st>>>(o.b2, g2, N2 * 64);

  o.b3 = allocF(ws, (size_t)N3 * 128);
  { size_t mk = ws.off;
    float* t = allocF(ws, (size_t)N3 * 64);
    conv_op(ws, st, o.b2, N2, 64, d23, 8, N3, e.w3, 64, 64, t);
    bn_op(ws, st, t, t, N3, 64, e.bn3, nullptr, 1);
    run_blocks(ws, st, o.b3, t, N3, k3l3, e.b3);
    ws.off = mk; }
  if (g3) k_sigmul<<<dim3(ceil_div(N3 * 128, 256)), dim3(256), 0, st>>>(o.b3, g3, N3 * 128);

  o.b4 = allocF(ws, (size_t)N4 * 256);
  { size_t mk = ws.off;
    float* t = allocF(ws, (size_t)N4 * 128);
    conv_op(ws, st, o.b3, N3, 128, d34, 8, N4, e.w4, 128, 128, t);
    bn_op(ws, st, t, t, N4, 128, e.bn4, nullptr, 1);
    run_blocks(ws, st, o.b4, t, N4, k3l4, e.b4);
    ws.off = mk; }
  return o;
}

static void upconv_op(WS& ws, hipStream_t st, const float* x, int N_c, int Cin,
                      const int* par, const int* off, int N_f,
                      const float* w, int Cout, const BNp& bnp, float* out){
  size_t mk = ws.off;
  int* nbr = (int*)ws.alloc((size_t)8 * N_f * sizeof(int));
  k_upnbr<<<dim3(ceil_div(N_f, 256)), dim3(256), 0, st>>>(par, off, N_f, nbr);
  conv_op(ws, st, x, N_c, Cin, nbr, 8, N_f, w, Cout, Cout, out);
  bn_op(ws, st, out, out, N_f, Cout, bnp, nullptr, 1);
  ws.off = mk;
}

extern "C" void kernel_launch(void* const* d_in, const int* in_sizes, int n_in,
                              void* d_out, int out_size, void* d_ws, size_t ws_size,
                              hipStream_t stream){
  (void)n_in; (void)out_size;
  Reader rd{d_in, in_sizes, 0};
  const float* feats_main    = rd.F();
  const float* feats_support = rd.F();
  int N0 = in_sizes[0] / 3;

  Enc senc = rdEnc(rd);
  Enc menc = rdEnc(rd);
  const float* wt2 = rd.F();
  const float* wt3 = rd.F();
  const float* wt4 = rd.F();
  const float* wmt = rd.F();
  Dec dec = rdDec(rd);

  // maps
  const int* k5l0 = rd.I();
  const int* k3l0 = rd.I();
  int N1 = in_sizes[rd.i] / 27; const int* k3l1 = rd.I();
  int N2 = in_sizes[rd.i] / 27; const int* k3l2 = rd.I();
  int N3 = in_sizes[rd.i] / 27; const int* k3l3 = rd.I();
  int N4 = in_sizes[rd.i] / 27; const int* k3l4 = rd.I();
  const int* d01 = rd.I();
  const int* d12 = rd.I();
  const int* d23 = rd.I();
  const int* d34 = rd.I();
  const int* up43p = rd.I(); const int* up43o = rd.I();
  const int* up32p = rd.I(); const int* up32o = rd.I();
  const int* up21p = rd.I(); const int* up21o = rd.I();
  const int* up10p = rd.I(); const int* up10o = rd.I();

  WS ws{(char*)d_ws, ws_size, 0};
  hipStream_t st = stream;

  // ---- support encoder (no gating) ----
  EncOut rs = run_encoder(ws, st, feats_support, senc, N0, N1, N2, N3, N4,
                          k5l0, d01, d12, d23, d34, k3l1, k3l2, k3l3, k3l4,
                          nullptr, nullptr, nullptr);

  // ---- surppot projections (plain matmuls) ----
  float* s2 = allocF(ws, (size_t)N1 * 32);
  conv_op(ws, st, rs.b1, N1, 32, nullptr, 1, N1, wt2, 32, 32, s2);
  float* s3 = allocF(ws, (size_t)N2 * 64);
  conv_op(ws, st, rs.b2, N2, 64, nullptr, 1, N2, wt3, 64, 64, s3);
  float* s4 = allocF(ws, (size_t)N3 * 128);
  conv_op(ws, st, rs.b3, N3, 128, nullptr, 1, N3, wt4, 128, 128, s4);

  // ---- main encoder with gates ----
  EncOut rm = run_encoder(ws, st, feats_main, menc, N0, N1, N2, N3, N4,
                          k5l0, d01, d12, d23, d34, k3l1, k3l2, k3l3, k3l4,
                          s2, s3, s4);

  // b4cat = concat(rm.b4, rs.b4)  [N4, 512]; then x4 = b4cat @ wmt  [N4, 256]
  float* b4cat = allocF(ws, (size_t)N4 * 512);
  k_copy_cols<<<dim3(ceil_div(N4 * 256, 256)), dim3(256), 0, st>>>(b4cat, 512, 0,   rm.b4, 256, N4);
  k_copy_cols<<<dim3(ceil_div(N4 * 256, 256)), dim3(256), 0, st>>>(b4cat, 512, 256, rs.b4, 256, N4);
  float* x4 = allocF(ws, (size_t)N4 * 256);
  conv_op(ws, st, b4cat, N4, 512, nullptr, 1, N4, wmt, 256, 256, x4);

  // ---- decoder ----
  float* o4 = allocF(ws, (size_t)N3 * 256);
  upconv_op(ws, st, x4, N4, 256, up43p, up43o, N3, dec.wtr4, 256, dec.bntr4, o4);
  float* c5 = allocF(ws, (size_t)N3 * 384);
  k_copy_cols<<<dim3(ceil_div(N3 * 256, 256)), dim3(256), 0, st>>>(c5, 384, 0,   o4,    256, N3);
  k_copy_cols<<<dim3(ceil_div(N3 * 128, 256)), dim3(256), 0, st>>>(c5, 384, 256, rm.b3, 128, N3);
  float* o5 = allocF(ws, (size_t)N3 * 256);
  run_blocks(ws, st, o5, c5, N3, k3l3, dec.b5);

  float* o6u = allocF(ws, (size_t)N2 * 128);
  upconv_op(ws, st, o5, N3, 256, up32p, up32o, N2, dec.wtr5, 128, dec.bntr5, o6u);
  float* c6 = allocF(ws, (size_t)N2 * 192);
  k_copy_cols<<<dim3(ceil_div(N2 * 128, 256)), dim3(256), 0, st>>>(c6, 192, 0,   o6u,   128, N2);
  k_copy_cols<<<dim3(ceil_div(N2 * 64, 256)), dim3(256), 0, st>>>(c6, 192, 128, rm.b2, 64,  N2);
  float* o6 = allocF(ws, (size_t)N2 * 128);
  run_blocks(ws, st, o6, c6, N2, k3l2, dec.b6);

  float* o7u = allocF(ws, (size_t)N1 * 96);
  upconv_op(ws, st, o6, N2, 128, up21p, up21o, N1, dec.wtr6, 96, dec.bntr6, o7u);
  float* c7 = allocF(ws, (size_t)N1 * 128);
  k_copy_cols<<<dim3(ceil_div(N1 * 96, 256)), dim3(256), 0, st>>>(c7, 128, 0,  o7u,   96, N1);
  k_copy_cols<<<dim3(ceil_div(N1 * 32, 256)), dim3(256), 0, st>>>(c7, 128, 96, rm.b1, 32, N1);
  float* o7 = allocF(ws, (size_t)N1 * 96);
  run_blocks(ws, st, o7, c7, N1, k3l1, dec.b7);

  float* o8u = allocF(ws, (size_t)N0 * 96);
  upconv_op(ws, st, o7, N1, 96, up10p, up10o, N0, dec.wtr7, 96, dec.bntr7, o8u);
  float* c8 = allocF(ws, (size_t)N0 * 128);
  k_copy_cols<<<dim3(ceil_div(N0 * 96, 256)), dim3(256), 0, st>>>(c8, 128, 0,  o8u,   96, N0);
  k_copy_cols<<<dim3(ceil_div(N0 * 32, 256)), dim3(256), 0, st>>>(c8, 128, 96, rm.p1, 32, N0);
  float* fea = allocF(ws, (size_t)N0 * 96);
  run_blocks(ws, st, fea, c8, N0, k3l0, dec.b8);

  // ---- final head: logits = fea @ wf + bf (Cout 20 padded to 32) ----
  size_t mk = ws.off;
  float* lin = allocF(ws, (size_t)N0 * 32);
  conv_op(ws, st, fea, N0, 96, nullptr, 1, N0, dec.wf, 20, 32, lin);
  k_finalout<<<dim3(ceil_div(N0 * 116, 256)), dim3(256), 0, st>>>(lin, 32, dec.bf, fea,
                                                                  (float*)d_out, N0);
  ws.off = mk;
}
